// GAT_66245575574016
// MI455X (gfx1250) — matrix operand order repaired and checked by reading, not compiled, not run on hardware
//
#include <hip/hip_runtime.h>

typedef __attribute__((ext_vector_type(16))) _Float16 v16h;
typedef __attribute__((ext_vector_type(8)))  _Float16 v8h;
typedef __attribute__((ext_vector_type(8)))  float    v8f;

#define NEG_SLOPE 0.2f
#define BS 256

__device__ __forceinline__ float lrelu(float v) { return v > 0.f ? v : NEG_SLOPE * v; }

// Order-preserving map  float -> uint  (for atomicMax-based segment_max)
__device__ __forceinline__ unsigned fenc(float f) {
  unsigned b = __float_as_uint(f);
  return (b & 0x80000000u) ? ~b : (b | 0x80000000u);
}
__device__ __forceinline__ float fdec(unsigned u) {
  unsigned b = (u & 0x80000000u) ? (u & 0x7FFFFFFFu) : ~u;
  return __uint_as_float(b);
}
// where(isfinite(m), m, 0): untouched slots decode to NaN -> 0 (matches reference)
__device__ __forceinline__ float fin0(float m) {
  return (__builtin_fabsf(m) < __builtin_inff()) ? m : 0.f;
}

// ---------------------------------------------------------------- utilities
__global__ void zero_f32(float* __restrict__ p, long n) {
  long i = (long)blockIdx.x * blockDim.x + threadIdx.x;
  if (i < n) p[i] = 0.f;
}

// ---------------------------------------------------------------- WMMA GEMM
// C[M,Nc] = A[M,K] @ B[K,Nc]; fp32 in memory, f16 WMMA with f32 accumulate.
// One block per 16-row stripe; tilesN (= Nc/16) waves per block, one 16x16 C
// tile per wave. A and B tiles are staged through LDS:
//   Alds[16][32]  f16 row-major        (A tile loaded once per block, not per wave)
//   Bt  [Nc][32]  f16 K-contiguous     (coalesced global reads, transposed store)
// so each lane's WMMA operands are contiguous 16B ds_load_b128s.
__global__ void gemm_wmma_f16(const float* __restrict__ A, const float* __restrict__ B,
                              float* __restrict__ C, int K, int Nc) {
  extern __shared__ _Float16 smem[];
  _Float16* Alds = smem;             // 16*32 halves
  _Float16* Bt   = smem + 16 * 32;   // Nc*32 halves

  const int tid   = threadIdx.x;
  const int nthr  = blockDim.x;      // 32 * tilesN
  const int tM    = blockIdx.x;      // 16-row stripe
  const int tN    = tid >> 5;        // wave -> 16-col tile
  const int lane  = tid & 31;
  const int r     = lane & 15;       // A row / B,C column within tile
  const int half  = lane >> 4;

  union { v16h v; v8h h[2]; } a, b;
  union { v8f  v; float e[8]; } c;
#pragma unroll
  for (int j = 0; j < 8; ++j) c.e[j] = 0.f;

  const int btile = 32 * Nc;

  for (int k0 = 0; k0 < K; k0 += 32) {
    // ---- stage A tile (16x32), coalesced along K ----
    for (int idx = tid; idx < 16 * 32; idx += nthr) {
      int row = idx >> 5, kk = idx & 31;
      Alds[row * 32 + kk] = (_Float16)A[(long)(tM * 16 + row) * K + k0 + kk];
    }
    // ---- stage B tile (32xNc) transposed, coalesced along N ----
    for (int idx = tid; idx < btile; idx += nthr) {
      int kk = idx / Nc, n = idx - kk * Nc;
      Bt[n * 32 + kk] = (_Float16)B[(long)(k0 + kk) * Nc + n];
    }
    __syncthreads();

    // ISA 16-bit A 16x32 layout: lanes<16: K=0..7,16..23 ; lanes>=16: K=8..15,24..31
    a.h[0] = *(const v8h*)&Alds[r * 32 + half * 8];
    a.h[1] = *(const v8h*)&Alds[r * 32 + half * 8 + 16];
    // ISA 16-bit B 32x16 layout: lanes<16: K=0..7,16..23 ; lanes>=16: K=8..15,24..31
    const _Float16* bp = &Bt[(tN * 16 + r) * 32 + half * 8];
    b.h[0] = *(const v8h*)bp;
    b.h[1] = *(const v8h*)(bp + 16);

    // (neg_a, A, neg_b, B, c_mod, C, reuse_a, reuse_b)
    c.v = __builtin_amdgcn_wmma_f32_16x16x32_f16(false, a.v, false, b.v,
                                                 (short)0, c.v, false, false);
    __syncthreads();
  }

#pragma unroll
  for (int j = 0; j < 8; ++j)   // C/D layout: VGPR j -> M = j + 8*half, N = r
    C[(long)(tM * 16 + j + 8 * half) * Nc + tN * 16 + r] = c.e[j];
}

// ---------------------------------------------------------------- layer 1 attention
__global__ void l1_coef(const float* __restrict__ h1, const float* __restrict__ att_s,
                        const float* __restrict__ att_d, float* __restrict__ as,
                        float* __restrict__ ad, long NH) {
  long i = (long)blockIdx.x * blockDim.x + threadIdx.x;
  if (i >= NH) return;
  long n = i >> 3; int hd = (int)(i & 7);
  const float* hp = h1 + n * 128 + hd * 16;
  const float* ws = att_s + hd * 16;
  const float* wd = att_d + hd * 16;
  float s = 0.f, d = 0.f;
#pragma unroll
  for (int c = 0; c < 16; ++c) { s += hp[c] * ws[c]; d += hp[c] * wd[c]; }
  as[i] = s; ad[i] = d;
}

__global__ void l1_edge_max(const int* __restrict__ src, const int* __restrict__ dst,
                            const float* __restrict__ as, const float* __restrict__ ad,
                            unsigned* __restrict__ menc, long EH) {
  long i = (long)blockIdx.x * blockDim.x + threadIdx.x;
  if (i >= EH) return;
  long e = i >> 3; int hd = (int)(i & 7);
  int s = src[e], d = dst[e];
  float v = lrelu(as[(long)s * 8 + hd] + ad[(long)d * 8 + hd]);
  atomicMax(&menc[(long)d * 8 + hd], fenc(v));
}

__global__ void l1_edge_sum(const int* __restrict__ src, const int* __restrict__ dst,
                            const float* __restrict__ as, const float* __restrict__ ad,
                            const unsigned* __restrict__ menc, float* __restrict__ denom,
                            long EH) {
  long i = (long)blockIdx.x * blockDim.x + threadIdx.x;
  if (i >= EH) return;
  long e = i >> 3; int hd = (int)(i & 7);
  int s = src[e], d = dst[e];
  float v = lrelu(as[(long)s * 8 + hd] + ad[(long)d * 8 + hd]);
  float m = fin0(fdec(menc[(long)d * 8 + hd]));
  atomicAdd(&denom[(long)d * 8 + hd], __expf(v - m));
}

__global__ void l1_scatter(const int* __restrict__ src, const int* __restrict__ dst,
                           const float* __restrict__ as, const float* __restrict__ ad,
                           const unsigned* __restrict__ menc, const float* __restrict__ denom,
                           const float* __restrict__ h1, float* __restrict__ out1, long EH) {
  long i = (long)blockIdx.x * blockDim.x + threadIdx.x;
  if (i >= EH) return;
  long e = i >> 3; int hd = (int)(i & 7);
  int s = src[e], d = dst[e];
  float v = lrelu(as[(long)s * 8 + hd] + ad[(long)d * 8 + hd]);
  float m = fin0(fdec(menc[(long)d * 8 + hd]));
  float alpha = __expf(v - m) / (denom[(long)d * 8 + hd] + 1e-16f);
  const float* hp = h1 + (long)s * 128 + hd * 16;
  float* op = out1 + (long)d * 128 + hd * 16;
#pragma unroll
  for (int c = 0; c < 16; ++c) atomicAdd(&op[c], hp[c] * alpha);
}

// bias + ELU in place (concat output of layer 1)
__global__ void l1_finish(float* __restrict__ out1, const float* __restrict__ b1, long n128) {
  long i = (long)blockIdx.x * blockDim.x + threadIdx.x;
  if (i >= n128) return;
  float v = out1[i] + b1[i & 127];
  out1[i] = v > 0.f ? v : expm1f(v);
}

// ---------------------------------------------------------------- layer 2 attention
__global__ void l2_coef(const float* __restrict__ h2, const float* __restrict__ att_s,
                        const float* __restrict__ att_d, float* __restrict__ as,
                        float* __restrict__ ad, long N) {
  long n = (long)blockIdx.x * blockDim.x + threadIdx.x;
  if (n >= N) return;
  const float* hp = h2 + n * 64;
  float s = 0.f, d = 0.f;
#pragma unroll 8
  for (int c = 0; c < 64; ++c) { s += hp[c] * att_s[c]; d += hp[c] * att_d[c]; }
  as[n] = s; ad[n] = d;
}

__global__ void l2_edge_max(const int* __restrict__ src, const int* __restrict__ dst,
                            const float* __restrict__ as, const float* __restrict__ ad,
                            unsigned* __restrict__ menc, long E) {
  long e = (long)blockIdx.x * blockDim.x + threadIdx.x;
  if (e >= E) return;
  float v = lrelu(as[src[e]] + ad[dst[e]]);
  atomicMax(&menc[dst[e]], fenc(v));
}

__global__ void l2_edge_sum(const int* __restrict__ src, const int* __restrict__ dst,
                            const float* __restrict__ as, const float* __restrict__ ad,
                            const unsigned* __restrict__ menc, float* __restrict__ denom,
                            float* __restrict__ expv, long E) {
  long e = (long)blockIdx.x * blockDim.x + threadIdx.x;
  if (e >= E) return;
  int d = dst[e];
  float v = lrelu(as[src[e]] + ad[d]);
  float ev = __expf(v - fin0(fdec(menc[d])));
  expv[e] = ev;
  atomicAdd(&denom[d], ev);
}

__global__ void l2_scatter(const int* __restrict__ src, const int* __restrict__ dst,
                           const float* __restrict__ h2, const float* __restrict__ expv,
                           const float* __restrict__ denom, float* __restrict__ out, long EC) {
  long i = (long)blockIdx.x * blockDim.x + threadIdx.x;
  if (i >= EC) return;
  long e = i >> 6; int c = (int)(i & 63);
  int s = src[e], d = dst[e];
  float alpha = expv[e] / (denom[d] + 1e-16f);
  atomicAdd(&out[(long)d * 64 + c], h2[(long)s * 64 + c] * alpha);
}

// bias + log_softmax in place over 64 classes
__global__ void finalize(float* __restrict__ out, const float* __restrict__ b2, long N) {
  long n = (long)blockIdx.x * blockDim.x + threadIdx.x;
  if (n >= N) return;
  float* row = out + n * 64;
  float mx = -__builtin_inff();
#pragma unroll 8
  for (int c = 0; c < 64; ++c) { float v = row[c] + b2[c]; mx = v > mx ? v : mx; }
  float s = 0.f;
#pragma unroll 8
  for (int c = 0; c < 64; ++c) s += __expf(row[c] + b2[c] - mx);
  float ls = mx + __logf(s);
#pragma unroll 8
  for (int c = 0; c < 64; ++c) row[c] = row[c] + b2[c] - ls;
}

// ---------------------------------------------------------------- host
static inline long cdiv(long a, long b) { return (a + b - 1) / b; }

extern "C" void kernel_launch(void* const* d_in, const int* in_sizes, int n_in,
                              void* d_out, int out_size, void* d_ws, size_t ws_size,
                              hipStream_t stream) {
  const float* x      = (const float*)d_in[0];
  const int*   ei     = (const int*)d_in[1];
  const float* W1     = (const float*)d_in[2];
  const float* att1_s = (const float*)d_in[3];
  const float* att1_d = (const float*)d_in[4];
  const float* b1     = (const float*)d_in[5];
  const float* W2     = (const float*)d_in[6];
  const float* att2_s = (const float*)d_in[7];
  const float* att2_d = (const float*)d_in[8];
  const float* b2     = (const float*)d_in[9];
  float* out = (float*)d_out;

  const long N = in_sizes[0] / 256;   // 100000
  const long E = in_sizes[1] / 2;     // 1600000
  const int* src = ei;
  const int* dst = ei + E;

  // workspace layout (floats)
  float* f = (float*)d_ws;
  float*    h1    = f;               f += N * 128;
  float*    out1  = f;               f += N * 128;
  float*    a1s   = f;               f += N * 8;
  float*    a1d   = f;               f += N * 8;
  unsigned* m1    = (unsigned*)f;    f += N * 8;
  float*    den1  = f;               f += N * 8;
  float*    h2    = f;               f += N * 64;
  float*    a2s   = f;               f += N;
  float*    a2d   = f;               f += N;
  unsigned* m2    = (unsigned*)f;    f += N;
  float*    den2  = f;               f += N;
  float*    expv2 = f;               f += E;

  // zero accumulators (enc==0 decodes to NaN -> "-inf then 0", as in reference)
  zero_f32<<<cdiv(N * 128, BS), BS, 0, stream>>>(out1, N * 128);
  zero_f32<<<cdiv(N * 8,  BS), BS, 0, stream>>>((float*)m1, N * 8);
  zero_f32<<<cdiv(N * 8,  BS), BS, 0, stream>>>(den1, N * 8);
  zero_f32<<<cdiv(N,      BS), BS, 0, stream>>>((float*)m2, N);
  zero_f32<<<cdiv(N,      BS), BS, 0, stream>>>(den2, N);
  zero_f32<<<cdiv(N * 64, BS), BS, 0, stream>>>(out, N * 64);

  // ---- layer 1: h1 = x @ W1  (K=256, Nc=128; 8 waves/block, 9KB LDS) ----
  {
    size_t shmem = (size_t)(16 * 32 + 32 * 128) * sizeof(_Float16);
    gemm_wmma_f16<<<N / 16, 8 * 32, shmem, stream>>>(x, W1, h1, 256, 128);
  }
  l1_coef<<<cdiv(N * 8, BS), BS, 0, stream>>>(h1, att1_s, att1_d, a1s, a1d, N * 8);

  const long EH = E * 8;
  l1_edge_max<<<cdiv(EH, BS), BS, 0, stream>>>(src, dst, a1s, a1d, m1, EH);
  l1_edge_sum<<<cdiv(EH, BS), BS, 0, stream>>>(src, dst, a1s, a1d, m1, den1, EH);
  l1_scatter <<<cdiv(EH, BS), BS, 0, stream>>>(src, dst, a1s, a1d, m1, den1, h1, out1, EH);
  l1_finish  <<<cdiv(N * 128, BS), BS, 0, stream>>>(out1, b1, N * 128);

  // ---- layer 2: h2 = ELU(out1) @ W2  (K=128, Nc=64; 4 waves/block, 5KB LDS) ----
  {
    size_t shmem = (size_t)(16 * 32 + 32 * 64) * sizeof(_Float16);
    gemm_wmma_f16<<<N / 16, 4 * 32, shmem, stream>>>(out1, W2, h2, 128, 64);
  }
  l2_coef<<<cdiv(N, BS), BS, 0, stream>>>(h2, att2_s, att2_d, a2s, a2d, N);

  l2_edge_max<<<cdiv(E, BS), BS, 0, stream>>>(src, dst, a2s, a2d, m2, E);
  l2_edge_sum<<<cdiv(E, BS), BS, 0, stream>>>(src, dst, a2s, a2d, m2, den2, expv2, E);
  l2_scatter <<<cdiv(E * 64, BS), BS, 0, stream>>>(src, dst, h2, expv2, den2, out, E * 64);
  finalize   <<<cdiv(N, BS), BS, 0, stream>>>(out, b2, N);
}
